// NonLocalSpatialPropagation_65541200937382
// MI455X (gfx1250) — compile-verified, matrix operand order checked by reading, not run on hardware
//
#include <hip/hip_runtime.h>
#include <hip/hip_bf16.h>
#include <math.h>

typedef __attribute__((ext_vector_type(16))) _Float16 v16h;
typedef __attribute__((ext_vector_type(8)))  _Float16 v8h;
typedef __attribute__((ext_vector_type(8)))  float    v8f;

#define BATCH 4
#define HH 256
#define WW 512
#define HWSZ (HH*WW)
#define HP (HH+2)
#define WP (WW+2)
static constexpr size_t NPIX  = (size_t)BATCH * HWSZ;
static constexpr size_t NPIXP = (size_t)BATCH * HP * WP; // haloed pixel count

static constexpr size_t ALN(size_t x) { return (x + 255) & ~(size_t)255; }

// ---------------- workspace layout (bytes) ----------------
static constexpr size_t OFF_WF    = 0;                         // weight fragments (f16) < 128KB
static constexpr size_t OFF_STAT  = 128 * 1024;                // bn stats
static constexpr size_t OFF_LDOWN = 256 * 1024;                // f32 [4][3][H][W]
static constexpr size_t OFF_RDOWN = ALN(OFF_LDOWN + 3 * NPIX * 4);
static constexpr size_t OFF_GUID  = ALN(OFF_RDOWN + 3 * NPIX * 4);   // f16 NHWC halo, Cpad=16
static constexpr size_t OFF_C1RAW = ALN(OFF_GUID  + NPIXP * 16 * 2); // f32 NCHW [4][32][H][W]
static constexpr size_t OFF_H1    = ALN(OFF_C1RAW + 32 * NPIX * 4);  // f16 NHWC halo, C=32
static constexpr size_t OFF_C2RAW = ALN(OFF_H1    + NPIXP * 32 * 2); // f32 NCHW [4][64][H][W]
static constexpr size_t OFF_H2    = ALN(OFF_C2RAW + 64 * NPIX * 4);  // f16 NHWC halo, C=64
static constexpr size_t OFF_OA    = ALN(OFF_H2    + NPIXP * 64 * 2); // f32 NCHW [4][24][H][W]

// weight-fragment sub-offsets (halfs): conv1 MT=2,KT=5 -> 5120; conv2 MT=4,KT=9 -> 18432; conv3 MT=2,KT=18 -> 18432
static constexpr size_t W1F = 0, W2F = 5120, W3F = 23552;

// ---------------- weight repack into WMMA A-fragment layout ----------------
// dst[((mt*KT + kt)*32 + lane)*16 + e]; row m = mt*16 + (lane&15)
// k = kt*32 + (e&7) + 8*(lane>>4) + 16*(e>>3); ordering k = cin + Cpad*tap
__global__ void k_prep_w(const float* __restrict__ w, _Float16* __restrict__ dst,
                         int Cout, int CinReal, int Cpad, int MT, int KT) {
  int i = blockIdx.x * 256 + threadIdx.x;
  int total = MT * KT * 32 * 16;
  if (i >= total) return;
  int e     = i & 15;
  int rest  = i >> 4;
  int lane  = rest & 31;
  int rest2 = rest >> 5;
  int kt = rest2 % KT;
  int mt = rest2 / KT;
  int m   = mt * 16 + (lane & 15);
  int khi = lane >> 4;
  int k   = kt * 32 + (e & 7) + (khi << 3) + ((e >> 3) << 4);
  int K   = Cpad * 9;
  float v = 0.f;
  if (m < Cout && k < K) {
    int cin = k % Cpad;
    int tap = k / Cpad;
    if (cin < CinReal) {
      int ky = tap / 3, kx = tap % 3;
      v = w[((m * CinReal + cin) * 3 + ky) * 3 + kx];
    }
  }
  dst[i] = (_Float16)v;
}

// ---------------- vector zero (16B granules) ----------------
__global__ void k_zero16(uint4* __restrict__ p, size_t n16) {
  size_t i = (size_t)blockIdx.x * 256 + threadIdx.x;
  if (i < n16) p[i] = make_uint4(0u, 0u, 0u, 0u);
}
__global__ void k_zero(float* p, int nelem) {
  int i = blockIdx.x * 256 + threadIdx.x;
  if (i < nelem) p[i] = 0.f;
}

// ---------------- 2x bilinear (antialiased triangle) downsample ----------------
__global__ void k_down2(const float* __restrict__ src, float* __restrict__ dst, int C) {
  size_t i = (size_t)blockIdx.x * 256 + threadIdx.x;
  size_t total = (size_t)BATCH * C * HWSZ;
  if (i >= total) return;
  int x = i % WW;
  int y = (i / WW) % HH;
  int bc = i / HWSZ;
  const float* s = src + (size_t)bc * 4 * HWSZ; // src plane is (2H)x(2W)
  const int H2 = 2 * HH, W2 = 2 * WW;
  const float wt[4] = {1.f, 3.f, 3.f, 1.f};
  float acc = 0.f, wsum = 0.f;
  for (int iy = 0; iy < 4; ++iy) {
    int j = 2 * y - 1 + iy;
    if (j < 0 || j >= H2) continue;
    for (int ix = 0; ix < 4; ++ix) {
      int q = 2 * x - 1 + ix;
      if (q < 0 || q >= W2) continue;
      float wgt = wt[iy] * wt[ix];
      acc += wgt * s[(size_t)j * W2 + q];
      wsum += wgt;
    }
  }
  dst[i] = acc / wsum;
}

// ---------------- guidance = [normal, L, R, warped(R)-L] -> f16 NHWC halo (Cpad=16) ----------------
__global__ void k_guidance(const float* __restrict__ normal, const float* __restrict__ coarse,
                           const float* __restrict__ Ld, const float* __restrict__ Rd,
                           _Float16* __restrict__ guid) {
  size_t i = (size_t)blockIdx.x * 256 + threadIdx.x;
  if (i >= NPIX) return;
  int x = i % WW;
  int y = (i / WW) % HH;
  int b = i / HWSZ;
  size_t p = (size_t)y * WW + x;
  float disp = coarse[(size_t)b * HWSZ + p] * 0.5f; // scale = 2
  float xs = (float)x - disp;
  float x0f = floorf(xs);
  int xq = (int)x0f;
  float wx = xs - x0f;
  _Float16* g = guid + ((size_t)(b * HP + y + 1) * WP + (x + 1)) * 16;
  for (int c = 0; c < 3; ++c) {
    size_t plane = (size_t)(b * 3 + c) * HWSZ;
    float L = Ld[plane + p];
    float R = Rd[plane + p];
    const float* Rrow = Rd + plane + (size_t)y * WW;
    float t0 = (xq >= 0 && xq < WW) ? Rrow[xq] : 0.f;
    float t1 = (xq + 1 >= 0 && xq + 1 < WW) ? Rrow[xq + 1] : 0.f;
    float wv = t0 * (1.f - wx) + t1 * wx; // y-weight is exactly 0 (ys integral)
    g[0 + c] = (_Float16)normal[plane + p];
    g[3 + c] = (_Float16)L;
    g[6 + c] = (_Float16)R;
    g[9 + c] = (_Float16)(wv - L);
  }
}

// ---------------- implicit-GEMM 3x3 conv via v_wmma_f32_16x16x32_f16 ----------------
// activations: f16 NHWC with 1-px zero halo (Cpad channels); one wave = 16 pixels x Cout
template <int Cpad, int Cout, int MT, int KT>
__global__ __launch_bounds__(128) void k_conv(const _Float16* __restrict__ in,
                                              const _Float16* __restrict__ wf,
                                              float* __restrict__ out, int ntiles) {
  const int lane = threadIdx.x & 31;
  int t = blockIdx.x * 4 + (threadIdx.x >> 5);
  if (t >= ntiles) return; // uniform per wave
  const int tilesPerRow = WW >> 4;
  int b = t / (HH * tilesPerRow);
  int r = t - b * (HH * tilesPerRow);
  int y  = r / tilesPerRow;
  int x0 = (r - y * tilesPerRow) << 4;
  const int n   = lane & 15;
  const int khi = lane >> 4;
  const int x   = x0 + n;
  constexpr int K = Cpad * 9;

  v8f zero = {};
  v8f acc[MT];
#pragma unroll
  for (int m = 0; m < MT; ++m) acc[m] = zero;

  const v16h* wfv = (const v16h*)wf;
  const _Float16* inb = in + ((size_t)(b * HP + y + 1) * WP + (x + 1)) * Cpad;

#pragma unroll
  for (int kt = 0; kt < KT; ++kt) {
    v8h run0 = {}, run1 = {};
    {
      // run 0: elements 0..7 -> k = kt*32 + khi*8 + [0..7]
      int kb = kt * 32; // + khi*8 (khi does not change tap/ validity: K % 16 == 0)
      if (kb < K) {
        int tap = (kt * 32) / Cpad;           // compile-time per kt
        int cin0 = (kt * 32) % Cpad + khi * 8;
        int dy = tap / 3 - 1, dx = tap % 3 - 1;
        int off = (dy * WP + dx) * Cpad + cin0;
        run0 = *(const v8h*)(inb + off);
      }
    }
    {
      // run 1: elements 8..15 -> k = kt*32 + 16 + khi*8 + [0..7]
      int kb = kt * 32 + 16;
      if (kb < K) {
        int tap = (kt * 32 + 16) / Cpad;      // compile-time per kt
        int cin0 = (kt * 32 + 16) % Cpad + khi * 8;
        int dy = tap / 3 - 1, dx = tap % 3 - 1;
        int off = (dy * WP + dx) * Cpad + cin0;
        run1 = *(const v8h*)(inb + off);
      }
    }
    v16h bf = __builtin_shufflevector(run0, run1, 0, 1, 2, 3, 4, 5, 6, 7,
                                      8, 9, 10, 11, 12, 13, 14, 15);
#pragma unroll
    for (int mt = 0; mt < MT; ++mt) {
      v16h af = wfv[(size_t)(mt * KT + kt) * 32 + lane];
      acc[mt] = __builtin_amdgcn_wmma_f32_16x16x32_f16(false, af, false, bf,
                                                       (short)0, acc[mt], false, false);
    }
  }

  float* outb = out + (size_t)b * Cout * HWSZ + (size_t)y * WW + x;
#pragma unroll
  for (int mt = 0; mt < MT; ++mt) {
#pragma unroll
    for (int rr = 0; rr < 8; ++rr) {
      int m = mt * 16 + rr + (khi << 3);
      if (m < Cout) outb[(size_t)m * HWSZ] = acc[mt][rr];
    }
  }
}

// ---------------- batchnorm statistics ----------------
#define STAT_CHUNKS 64
__global__ void k_stats(const float* __restrict__ x, float* __restrict__ stats, int C) {
  __shared__ float ssum[256];
  __shared__ float ssq[256];
  int c = blockIdx.x / STAT_CHUNKS;
  int chunk = blockIdx.x % STAT_CHUNKS;
  const int per = HWSZ / STAT_CHUNKS;
  float s = 0.f, q = 0.f;
  for (int b = 0; b < BATCH; ++b) {
    const float* xc = x + ((size_t)b * C + c) * HWSZ;
    for (int p = chunk * per + threadIdx.x; p < (chunk + 1) * per; p += 256) {
      float v = xc[p];
      s += v;
      q += v * v;
    }
  }
  ssum[threadIdx.x] = s;
  ssq[threadIdx.x] = q;
  __syncthreads();
  for (int st = 128; st > 0; st >>= 1) {
    if ((int)threadIdx.x < st) {
      ssum[threadIdx.x] += ssum[threadIdx.x + st];
      ssq[threadIdx.x]  += ssq[threadIdx.x + st];
    }
    __syncthreads();
  }
  if (threadIdx.x == 0) {
    atomicAdd(&stats[c], ssum[0]);
    atomicAdd(&stats[C + c], ssq[0]);
  }
}

// ---------------- BN + ReLU, NCHW f32 -> NHWC-halo f16 ----------------
template <int C>
__global__ void k_bn_relu(const float* __restrict__ x, const float* __restrict__ stats,
                          const float* __restrict__ g, const float* __restrict__ bb,
                          _Float16* __restrict__ out) {
  size_t i = (size_t)blockIdx.x * 256 + threadIdx.x;
  size_t total = (size_t)C * NPIX;
  if (i >= total) return;
  int c = (int)(i % C);
  size_t rest = i / C;
  int xx = (int)(rest % WW);
  int yy = (int)((rest / WW) % HH);
  int b  = (int)(rest / HWSZ);
  const float cnt = (float)BATCH * HWSZ;
  float m = stats[c] / cnt;
  float var = stats[C + c] / cnt - m * m;
  float xv = x[((size_t)b * C + c) * HWSZ + (size_t)yy * WW + xx];
  float v = (xv - m) * rsqrtf(var + 1e-5f) * g[c] + bb[c];
  out[((size_t)(b * HP + yy + 1) * WP + (xx + 1)) * C + c] = (_Float16)fmaxf(v, 0.f);
}

// ---------------- epilogue ----------------
__device__ __forceinline__ float tapv(const float* img, int yi, int xi) {
  if (yi < 0 || yi >= HH || xi < 0 || xi >= WW) return 0.f;
  return img[(size_t)yi * WW + xi];
}
__device__ __forceinline__ float bilin(const float* img, float ys, float xs) {
  float y0f = floorf(ys), x0f = floorf(xs);
  float wy = ys - y0f, wx = xs - x0f;
  int y0 = (int)y0f, x0 = (int)x0f;
  return tapv(img, y0, x0) * (1.f - wy) * (1.f - wx) +
         tapv(img, y0, x0 + 1) * (1.f - wy) * wx +
         tapv(img, y0 + 1, x0) * wy * (1.f - wx) +
         tapv(img, y0 + 1, x0 + 1) * wy * wx;
}

__global__ void k_finalize(const float* __restrict__ coarse, const float* __restrict__ conf,
                           const float* __restrict__ oa, const float* __restrict__ aff_scale,
                           float* __restrict__ out) {
  size_t i = (size_t)blockIdx.x * 256 + threadIdx.x;
  if (i >= NPIX) return;
  int x = i % WW;
  int y = (i / WW) % HH;
  int b = i / HWSZ;
  const float* oab = oa + (size_t)b * 24 * HWSZ + (size_t)y * WW + x;
  float o[24];
#pragma unroll
  for (int c = 0; c < 24; ++c) o[c] = oab[(size_t)c * HWSZ];

  float offy9[9], offx9[9];
#pragma unroll
  for (int n2 = 0; n2 < 4; ++n2) { offy9[n2] = o[2 * n2]; offx9[n2] = o[2 * n2 + 1]; }
  offy9[4] = 0.f; offx9[4] = 0.f;
#pragma unroll
  for (int n2 = 4; n2 < 8; ++n2) { offy9[n2 + 1] = o[2 * n2]; offx9[n2 + 1] = o[2 * n2 + 1]; }

  float asc = aff_scale[0] + 1e-8f;
  const float* confb = conf + (size_t)b * HWSZ;
  const float* crsb  = coarse + (size_t)b * HWSZ;

  float aff[8];
  float ssum = 0.f;
#pragma unroll
  for (int j = 0; j < 8; ++j) {
    int k9 = (j < 4) ? j : j + 1;
    float a = tanhf(o[16 + j]) / asc;
    float cf = bilin(confb, (float)y + offy9[k9], (float)x + offx9[k9]);
    aff[j] = a * cf;
    ssum += fabsf(aff[j]);
  }
  float s = fmaxf(ssum + 1e-4f, 1.0f);
  float tot = 0.f;
#pragma unroll
  for (int j = 0; j < 8; ++j) { aff[j] /= s; tot += aff[j]; }
  float aref = 1.0f - tot;
  float aff9[9];
#pragma unroll
  for (int j = 0; j < 4; ++j) aff9[j] = aff[j];
  aff9[4] = aref;
#pragma unroll
  for (int j = 4; j < 8; ++j) aff9[j + 1] = aff[j];

  float inter = 0.f;
#pragma unroll
  for (int k = 0; k < 9; ++k) {
    float ky = (float)(k / 3 - 1);
    float kx = (float)(k % 3 - 1);
    float sp = bilin(crsb, (float)y + ky + offy9[k], (float)x + kx + offx9[k]);
    inter += aff9[k] * sp;
  }
  inter = fmaxf(inter, 0.f);
  float center = aref;
  float others = fminf(fmaxf(1.f - center, 0.f), 1.f);
  float fin = center * crsb[(size_t)y * WW + x] + others * inter;
  out[i] = fmaxf(fin, 0.f);
}

// ---------------- launcher ----------------
extern "C" void kernel_launch(void* const* d_in, const int* in_sizes, int n_in,
                              void* d_out, int out_size, void* d_ws, size_t ws_size,
                              hipStream_t stream) {
  const float* coarse = (const float*)d_in[0];
  const float* normal = (const float*)d_in[1];
  const float* left   = (const float*)d_in[2];
  const float* right  = (const float*)d_in[3];
  const float* conf   = (const float*)d_in[4];
  const float* w1 = (const float*)d_in[5];
  const float* g1 = (const float*)d_in[6];
  const float* b1 = (const float*)d_in[7];
  const float* w2 = (const float*)d_in[8];
  const float* g2 = (const float*)d_in[9];
  const float* b2 = (const float*)d_in[10];
  const float* w3 = (const float*)d_in[11];
  const float* ascale = (const float*)d_in[12];
  float* out = (float*)d_out;

  char* ws = (char*)d_ws;
  _Float16* wfrag = (_Float16*)(ws + OFF_WF);
  float*    stats = (float*)(ws + OFF_STAT);
  float*    Ld    = (float*)(ws + OFF_LDOWN);
  float*    Rd    = (float*)(ws + OFF_RDOWN);
  _Float16* guid  = (_Float16*)(ws + OFF_GUID);
  float*    c1raw = (float*)(ws + OFF_C1RAW);
  _Float16* h1    = (_Float16*)(ws + OFF_H1);
  float*    c2raw = (float*)(ws + OFF_C2RAW);
  _Float16* h2    = (_Float16*)(ws + OFF_H2);
  float*    oabuf = (float*)(ws + OFF_OA);

  // 0) zero the haloed NHWC activation buffers (halo + pad channels)
  {
    size_t n16g = NPIXP * 16 * 2 / 16;
    size_t n16a = NPIXP * 32 * 2 / 16;
    size_t n16b = NPIXP * 64 * 2 / 16;
    k_zero16<<<(int)((n16g + 255) / 256), 256, 0, stream>>>((uint4*)guid, n16g);
    k_zero16<<<(int)((n16a + 255) / 256), 256, 0, stream>>>((uint4*)h1, n16a);
    k_zero16<<<(int)((n16b + 255) / 256), 256, 0, stream>>>((uint4*)h2, n16b);
  }

  // 1) repack conv weights into WMMA fragment layout (f16)
  k_prep_w<<<(2 * 5 * 512 + 255) / 256, 256, 0, stream>>>(w1, wfrag + W1F, 32, 12, 16, 2, 5);
  k_prep_w<<<(4 * 9 * 512 + 255) / 256, 256, 0, stream>>>(w2, wfrag + W2F, 64, 32, 32, 4, 9);
  k_prep_w<<<(2 * 18 * 512 + 255) / 256, 256, 0, stream>>>(w3, wfrag + W3F, 24, 64, 64, 2, 18);

  // 2) 2x downsample of left/right
  {
    size_t total = (size_t)BATCH * 3 * HWSZ;
    int blocks = (int)((total + 255) / 256);
    k_down2<<<blocks, 256, 0, stream>>>(left, Ld, 3);
    k_down2<<<blocks, 256, 0, stream>>>(right, Rd, 3);
  }

  // 3) guidance tensor (f16 NHWC halo, Cpad=16)
  k_guidance<<<(int)((NPIX + 255) / 256), 256, 0, stream>>>(normal, coarse, Ld, Rd, guid);

  const int ntiles = BATCH * HH * (WW >> 4); // 32768
  const int convBlocks = ntiles / 4;         // 4 waves per 128-thread block

  // 4) conv1 (16pad -> 32), bn-relu -> h1 (NHWC halo)
  k_conv<16, 32, 2, 5><<<convBlocks, 128, 0, stream>>>(guid, wfrag + W1F, c1raw, ntiles);
  k_zero<<<1, 256, 0, stream>>>(stats, 256);
  k_stats<<<32 * STAT_CHUNKS, 256, 0, stream>>>(c1raw, stats, 32);
  {
    size_t total = 32 * NPIX;
    k_bn_relu<32><<<(int)((total + 255) / 256), 256, 0, stream>>>(c1raw, stats, g1, b1, h1);
  }

  // 5) conv2 (32 -> 64), bn-relu -> h2 (NHWC halo)
  k_conv<32, 64, 4, 9><<<convBlocks, 128, 0, stream>>>(h1, wfrag + W2F, c2raw, ntiles);
  k_zero<<<1, 256, 0, stream>>>(stats, 256);
  k_stats<<<64 * STAT_CHUNKS, 256, 0, stream>>>(c2raw, stats, 64);
  {
    size_t total = 64 * NPIX;
    k_bn_relu<64><<<(int)((total + 255) / 256), 256, 0, stream>>>(c2raw, stats, g2, b2, h2);
  }

  // 6) conv3 (64 -> 24) -> oa (f32 NCHW)
  k_conv<64, 24, 2, 18><<<convBlocks, 128, 0, stream>>>(h2, wfrag + W3F, oabuf, ntiles);

  // 7) epilogue
  k_finalize<<<(int)((NPIX + 255) / 256), 256, 0, stream>>>(coarse, conf, oabuf, ascale, out);
}
